// PiTWithCoords_79748952752794
// MI455X (gfx1250) — matrix operand order repaired and checked by reading, not compiled, not run on hardware
//
#include <hip/hip_runtime.h>
#include <hip/hip_bf16.h>
#include <math.h>

#define DEVINL __device__ __forceinline__

typedef unsigned short u16;
typedef unsigned int   u32;
typedef unsigned long long u64;
typedef __attribute__((ext_vector_type(16))) __bf16 v16bf;
typedef __attribute__((ext_vector_type(8)))  float  v8f;
typedef __attribute__((ext_vector_type(4)))  u32    v4u;

union FragBF { v16bf v; v4u q[2]; };

// ---- CDNA5 async Global->LDS copy path (ASYNCcnt), with safe fallback ----
#if defined(__gfx1250__) && __has_builtin(__builtin_amdgcn_global_load_async_to_lds_b128)
#define HAS_ASYNC_COPY 1
typedef __attribute__((ext_vector_type(4))) int i32x4;
typedef __attribute__((ext_vector_type(2))) int i32x2;
typedef __attribute__((address_space(1))) i32x4 g_i32x4;   // global pointee types
typedef __attribute__((address_space(3))) i32x4 l_i32x4;   // LDS pointee types
typedef __attribute__((address_space(1))) i32x2 g_i32x2;
typedef __attribute__((address_space(3))) i32x2 l_i32x2;
#else
#define HAS_ASYNC_COPY 0
#endif

DEVINL void copy_g2l_16(const u16* g, u16* l) {   // 16 bytes
#if HAS_ASYNC_COPY
  __builtin_amdgcn_global_load_async_to_lds_b128(
      (g_i32x4*)const_cast<u16*>(g), (l_i32x4*)l, 0, 0);
#else
  *(v4u*)l = *(const v4u*)g;
#endif
}
DEVINL void copy_g2l_8(const u16* g, u16* l) {    // 8 bytes
#if HAS_ASYNC_COPY
  __builtin_amdgcn_global_load_async_to_lds_b64(
      (g_i32x2*)const_cast<u16*>(g), (l_i32x2*)l, 0, 0);
#else
  *(u64*)l = *(const u64*)g;
#endif
}

// Wait until at most N async copies remain in flight (in-order completion).
template <int N> DEVINL void async_join(bool keepN) {
#if HAS_ASYNC_COPY
#if __has_builtin(__builtin_amdgcn_s_wait_asynccnt)
  if (keepN) __builtin_amdgcn_s_wait_asynccnt(N);
  else       __builtin_amdgcn_s_wait_asynccnt(0);
#else
  if (keepN) asm volatile("s_wait_asynccnt %0" :: "n"(N) : "memory");
  else       asm volatile("s_wait_asynccnt 0" ::: "memory");
#endif
#else
  (void)keepN;
#endif
}

DEVINL u16 f32_to_bf16(float f) {
  u32 u = __float_as_uint(f);
  u32 lsb = (u >> 16) & 1u;
  u += 0x7fffu + lsb;                  // round-to-nearest-even
  return (u16)(u >> 16);
}
DEVINL float bf16_to_f32(u16 h) { return __uint_as_float(((u32)h) << 16); }

DEVINL float gelu_tanh(float x) {      // jax.nn.gelu(approximate=True)
  const float c0 = 0.7978845608028654f;
  float x3 = x * x * x;
  return 0.5f * x * (1.0f + tanhf(c0 * (x + 0.044715f * x3)));
}

// distance between point q of an rq x rq grid and point k of an rk x rk grid
DEVINL float grid_dist(int q, int rq, int k, int rk) {
  float xq = (float)(q % rq) / (float)rq;
  float yq = (float)(q / rq) / (float)rq;
  float xk = (float)(k % rk) / (float)rk;
  float yk = (float)(k / rk) / (float)rk;
  float dx = xq - xk, dy = yq - yk;
  return 0.5f * (dx * dx + dy * dy);
}

// ---------------------------------------------------------------------------
// Per-row percentile of the distance matrix (exact order statistics via
// binary search on the non-negative float bit pattern + linear interpolation).
// ---------------------------------------------------------------------------
__global__ __launch_bounds__(256) void row_percentile_kernel(
    float* __restrict__ thr, int rq, int rk, int NK, float pct) {
  __shared__ float srow[4096];
  __shared__ int   scnt[256];
  const int q = blockIdx.x;
  const int tid = threadIdx.x;
  for (int k = tid; k < NK; k += 256) srow[k] = grid_dist(q, rq, k, rk);
  __syncthreads();

  float idx = (float)(NK - 1) * pct * 0.01f;
  int r0 = (int)floorf(idx);
  float frac = idx - (float)r0;
  int r1 = (r0 + 1 < NK) ? (r0 + 1) : (NK - 1);

  float v[2];
  for (int which = 0; which < 2; ++which) {
    int rank = (which == 0) ? r0 : r1;
    u32 lo = 0u, hi = 0x7f800000u;
    while (lo < hi) {                       // uniform across threads
      u32 mid = lo + ((hi - lo) >> 1);
      float t = __uint_as_float(mid);
      int c = 0;
      for (int k = tid; k < NK; k += 256) c += (srow[k] <= t) ? 1 : 0;
      scnt[tid] = c; __syncthreads();
      for (int s = 128; s > 0; s >>= 1) {
        if (tid < s) scnt[tid] += scnt[tid + s];
        __syncthreads();
      }
      int total = scnt[0];
      __syncthreads();
      if (total >= rank + 1) hi = mid; else lo = mid + 1;
    }
    v[which] = __uint_as_float(hi);
  }
  if (tid == 0) thr[q] = v[0] * (1.0f - frac) + v[1] * frac;
}

// ---------------------------------------------------------------------------
// Build attention matrix att[h][q][k] (bf16), data-independent per launch:
//   scale = tan(pi/4*(1-1e-7)*(1+sin(r_h)));  sd = dist*scale
//   mask sd > thr_row*scale (if locality<=100), softmax(-sd) over k.
// ---------------------------------------------------------------------------
__global__ __launch_bounds__(256) void build_att_kernel(
    u16* __restrict__ att, const float* __restrict__ rvals,
    const float* __restrict__ thr, int rq, int rk, int Nq, int NK, int use_thr) {
  __shared__ float srow[4096];
  __shared__ float sred[256];
  const int q = blockIdx.x;
  const int h = blockIdx.y;
  const int tid = threadIdx.x;
  float r = rvals[h];
  float scale = tanf(0.25f * 3.14159265358979323846f * (1.0f - 1e-7f) * (1.0f + sinf(r)));
  float thr_s = use_thr ? thr[q] * scale : 3.0e38f;

  float lmin = 3.0e38f;
  for (int k = tid; k < NK; k += 256) {
    float sd = grid_dist(q, rq, k, rk) * scale;
    srow[k] = sd;
    lmin = fminf(lmin, sd);
  }
  sred[tid] = lmin; __syncthreads();
  for (int s = 128; s > 0; s >>= 1) {
    if (tid < s) sred[tid] = fminf(sred[tid], sred[tid + s]);
    __syncthreads();
  }
  float mn = sred[0]; __syncthreads();   // global min == min over kept entries

  float lsum = 0.0f;
  for (int k = tid; k < NK; k += 256) {
    float sd = srow[k];
    float e = (sd <= thr_s) ? __expf(mn - sd) : 0.0f;
    srow[k] = e;
    lsum += e;
  }
  sred[tid] = lsum; __syncthreads();
  for (int s = 128; s > 0; s >>= 1) {
    if (tid < s) sred[tid] += sred[tid + s];
    __syncthreads();
  }
  float inv = 1.0f / sred[0];

  u16* outp = att + ((size_t)h * Nq + q) * (size_t)NK;
  for (int k = tid; k < NK; k += 256) outp[k] = f32_to_bf16(srow[k] * inv);
}

// ---------------------------------------------------------------------------
// Conversion kernels
// ---------------------------------------------------------------------------
__global__ void f32_to_bf16_kernel(u16* __restrict__ o, const float* __restrict__ in, int n) {
  int i = blockIdx.x * 256 + threadIdx.x;
  if (i < n) o[i] = f32_to_bf16(in[i]);
}

// w[h][j][v] (8,256,32) f32 -> wt[(h*32+v)][j] (256,256) bf16   (B^T for WMMA)
__global__ void head_w_transpose_kernel(u16* __restrict__ o, const float* __restrict__ in) {
  int i = blockIdx.x * 256 + threadIdx.x;   // 65536
  int n = i >> 8;                           // h*32+v
  int j = i & 255;
  int h = n >> 5, v = n & 31;
  o[(size_t)n * 256 + j] = f32_to_bf16(in[(size_t)h * 8192 + (size_t)j * 32 + v]);
}

// Encoder: h = gelu(x @ W^T + b), K=4 (too small for WMMA) -> bf16
__global__ __launch_bounds__(256) void encoder_kernel(
    u16* __restrict__ hb, const float* __restrict__ x,
    const float* __restrict__ W, const float* __restrict__ b) {
  const int t = blockIdx.x;      // token in [0, B*NI)
  const int o = threadIdx.x;     // output channel
  float x0 = x[t * 4 + 0], x1 = x[t * 4 + 1], x2 = x[t * 4 + 2], x3 = x[t * 4 + 3];
  float acc = b[o] + x0 * W[o * 4 + 0] + x1 * W[o * 4 + 1]
                   + x2 * W[o * 4 + 2] + x3 * W[o * 4 + 3];
  hb[(size_t)t * 256 + o] = f32_to_bf16(gelu_tanh(acc));
}

// Decoder fc2: out[t] = dot256(h2[t], w) + b  (N=1, VALU + wave32 shuffle reduce)
__global__ __launch_bounds__(256) void decoder_fc2_kernel(
    float* __restrict__ outp, const u16* __restrict__ h2,
    const float* __restrict__ w, const float* __restrict__ b) {
  const int tid = threadIdx.x;
  const int wave = tid >> 5, lane = tid & 31;
  const int t = blockIdx.x * 8 + wave;
  const u16* row = h2 + (size_t)t * 256;
  float acc = 0.0f;
  for (int s = 0; s < 8; ++s) {
    int k = lane + s * 32;
    acc += bf16_to_f32(row[k]) * w[k];
  }
  for (int off = 16; off > 0; off >>= 1) acc += __shfl_down(acc, off, 32);
  if (lane == 0) outp[t] = acc + b[0];
}

// ---------------------------------------------------------------------------
// WMMA fragment loads (ISA 7.12.2 layouts). LDS rows padded to 40 bf16 (80 B).
// A (16x32, MxK): lane l -> row m=l%16, K in {8h..8h+7} U {16+8h..16+8h+7}
// B stored transposed [N][K]: lane l -> col n=l%16, K in {8h..8h+7} U {16+8h..16+8h+7}
// ---------------------------------------------------------------------------
DEVINL v16bf fragA(const u16* rowbase, int ln, int half) {
  const u16* p = rowbase + ln * 40 + 8 * half;
  FragBF f;
  f.q[0] = *(const v4u*)p;
  f.q[1] = *(const v4u*)(p + 16);
  return f.v;
}
DEVINL v16bf fragB(const u16* rowbase, int ln, int half) {
  const u16* p = rowbase + ln * 40 + 8 * half;
  FragBF f;
  f.q[0] = *(const v4u*)p;
  f.q[1] = *(const v4u*)(p + 16);
  return f.v;
}

#define F_BIAS 1
#define F_GELU 2
#define F_ADD  4
#define F_OUTF 8
#define F_OUTB 16

// MODE 0: C[m*N+n]                (plain lin layers)
// MODE 1: z=b*8+h; A per-head, Bt per-(b,h); C[(b*M+m)*256 + h*32 + n]
// MODE 2: value proj; bf16 store transposed to vt[(b*8+h)*32+v][k], k=m%Nk
template <int MODE, int BM, int BN, int WR, int WC, int FM, int FN>
__global__ __launch_bounds__(256) void gemm_bf16_kernel(
    const u16* __restrict__ A, const u16* __restrict__ Bt,
    float* __restrict__ outF, u16* __restrict__ outB,
    const float* __restrict__ bias, const float* __restrict__ addend,
    int M, int N, int K, int flags, int Nk) {
  static_assert(WR * WC == 8 && BM == WR * FM * 16 && BN == WC * FN * 16, "tile config");
  static_assert((BM * 4) % 256 == 0 && (BN == 64 || BN == 32), "stage config");
  constexpr int ASEG = (BM * 4) / 256;   // 16B async copies per thread for A tile
  constexpr int NPT  = ASEG + 1;         // total async issues per thread per tile

  __shared__ u16 sA[2][BM * 40];         // double-buffered K tiles
  __shared__ u16 sB[2][BN * 40];

  const int tid = threadIdx.x;
  const int wave = tid >> 5, lane = tid & 31;
  const int wr = wave / WC, wc = wave % WC;
  const int half = lane >> 4, ln = lane & 15;
  const int m0 = blockIdx.x * BM;
  const int n0 = blockIdx.y * BN;
  const int z = blockIdx.z;

  const u16* Ab = A;
  const u16* Bb = Bt;
  if (MODE == 1) {
    Ab = A + (size_t)(z & 7) * (size_t)M * (size_t)K;   // att shared across batch
    Bb = Bt + (size_t)z * (size_t)N * (size_t)K;        // vt per (b,h)
  }

  // compile-time unrolled staging coordinates (uniform async-op count per wave)
  int aRow[ASEG], aCol[ASEG];
#pragma unroll
  for (int s = 0; s < ASEG; ++s) {
    int seg = tid + s * 256;
    aRow[s] = seg >> 2;
    aCol[s] = (seg & 3) * 8;
  }
  int bRow, bCol;
  if (BN == 64) { bRow = tid >> 2; bCol = (tid & 3) * 8; }   // 16B per thread
  else          { bRow = tid >> 3; bCol = (tid & 7) * 4; }   // 8B per thread

  auto stage = [&](int buf, int kk) {
#pragma unroll
    for (int s = 0; s < ASEG; ++s)
      copy_g2l_16(&Ab[(size_t)(m0 + aRow[s]) * K + kk + aCol[s]],
                  &sA[buf][aRow[s] * 40 + aCol[s]]);
    if (BN == 64)
      copy_g2l_16(&Bb[(size_t)(n0 + bRow) * K + kk + bCol],
                  &sB[buf][bRow * 40 + bCol]);
    else
      copy_g2l_8(&Bb[(size_t)(n0 + bRow) * K + kk + bCol],
                 &sB[buf][bRow * 40 + bCol]);
  };

  v8f acc[FM][FN] = {};

  stage(0, 0);
  for (int kk = 0; kk < K; kk += 32) {
    const int cur = (kk >> 5) & 1;
    const bool more = (kk + 32) < K;
    if (more) {
      stage(cur ^ 1, kk + 32);                       // overlap next tile's copies
      if (kk + 64 < K)                               // hint tile after next
        __builtin_prefetch(&Ab[(size_t)(m0 + aRow[0]) * K + kk + 64], 0, 1);
    }
    async_join<NPT>(more);   // in-order: current tile's copies are complete
    __syncthreads();

    v16bf afr[FM], bfr[FN];
#pragma unroll
    for (int i = 0; i < FM; ++i)
      afr[i] = fragA(&sA[cur][(wr * FM * 16 + i * 16) * 40], ln, half);
#pragma unroll
    for (int j = 0; j < FN; ++j)
      bfr[j] = fragB(&sB[cur][(wc * FN * 16 + j * 16) * 40], ln, half);
#pragma unroll
    for (int i = 0; i < FM; ++i)
#pragma unroll
      for (int j = 0; j < FN; ++j)
        acc[i][j] = __builtin_amdgcn_wmma_f32_16x16x32_bf16(
            false, afr[i], false, bfr[j], (short)0, acc[i][j], false, false);
    __syncthreads();
  }

  // epilogue: C layout -> m = v + 8*half, n = lane%16 within each 16x16 frag
#pragma unroll
  for (int i = 0; i < FM; ++i) {
#pragma unroll
    for (int j = 0; j < FN; ++j) {
#pragma unroll
      for (int v = 0; v < 8; ++v) {
        int m = m0 + wr * FM * 16 + i * 16 + v + 8 * half;
        int n = n0 + wc * FN * 16 + j * 16 + ln;
        float xv = acc[i][j][v];
        if (flags & F_BIAS) xv += bias[n];
        if (flags & F_ADD)  xv += addend[(size_t)m * N + n];
        if (flags & F_GELU) xv = gelu_tanh(xv);
        if (MODE == 0) {
          size_t off = (size_t)m * N + n;
          if (flags & F_OUTF) outF[off] = xv;
          if (flags & F_OUTB) outB[off] = f32_to_bf16(xv);
        } else if (MODE == 1) {
          int b = z >> 3, h = z & 7;
          size_t off = ((size_t)b * M + m) * 256 + h * 32 + n;
          if (flags & F_OUTF) outF[off] = xv;
          if (flags & F_OUTB) outB[off] = f32_to_bf16(xv);
        } else {
          int b = m / Nk, kx = m - b * Nk;
          int hh = n >> 5, vv = n & 31;
          size_t off = ((size_t)(b * 8 + hh) * 32 + vv) * (size_t)Nk + kx;
          outB[off] = f32_to_bf16(xv);
        }
      }
    }
  }
}

// ---------------------------------------------------------------------------
extern "C" void kernel_launch(void* const* d_in, const int* in_sizes, int n_in,
                              void* d_out, int out_size, void* d_ws, size_t ws_size,
                              hipStream_t stream) {
  (void)in_sizes; (void)n_in; (void)out_size; (void)ws_size;
  const float* x     = (const float*)d_in[0];
  const float* enw   = (const float*)d_in[1];
  const float* enb   = (const float*)d_in[2];
  const float* downr = (const float*)d_in[3];
  const float* downw = (const float*)d_in[4];
  const float* par   = (const float*)d_in[5];
  const float* paw   = (const float*)d_in[6];
  const float* m1w   = (const float*)d_in[7];
  const float* m1b   = (const float*)d_in[8];
  const float* m2w   = (const float*)d_in[9];
  const float* m2b   = (const float*)d_in[10];
  const float* rsw   = (const float*)d_in[11];
  const float* rsb   = (const float*)d_in[12];
  const float* upr   = (const float*)d_in[13];
  const float* upw   = (const float*)d_in[14];
  const float* d1w   = (const float*)d_in[15];
  const float* d1b   = (const float*)d_in[16];
  const float* d2w   = (const float*)d_in[17];
  const float* d2b   = (const float*)d_in[18];
  float* outp = (float*)d_out;

  const int B = 8, NI = 4096, NL = 1024, H = 8, VD = 32, NB = 4;
  const int MBIG = B * NI;   // 32768
  const int MLAT = B * NL;   // 8192

  char* ws = (char*)d_ws;
  size_t cur = 0;
  auto carve = [&](size_t bytes) -> void* {
    void* p = (void*)(ws + cur);
    cur = (cur + bytes + 255) & ~(size_t)255;
    return p;
  };

  float* thrD = (float*)carve((size_t)NL * 4);
  float* thrP = (float*)carve((size_t)NL * 4);
  u16* attB   = (u16*)carve((size_t)H * NL * NI * 2);        // 64 MiB, reused
  u16* wtD    = (u16*)carve((size_t)65536 * 2);
  u16* wtU    = (u16*)carve((size_t)65536 * 2);
  u16* wtP    = (u16*)carve((size_t)NB * 65536 * 2);
  u16* w1B    = (u16*)carve((size_t)NB * 65536 * 2);
  u16* w2B    = (u16*)carve((size_t)NB * 65536 * 2);
  u16* wrB    = (u16*)carve((size_t)NB * 65536 * 2);
  u16* wd1B   = (u16*)carve((size_t)65536 * 2);
  u16* hbBig  = (u16*)carve((size_t)MBIG * 256 * 2);
  u16* hbLat  = (u16*)carve((size_t)MLAT * 256 * 2);
  u16* vtB    = (u16*)carve((size_t)B * H * VD * NI * 2);
  u16* paB    = (u16*)carve((size_t)MLAT * 256 * 2);
  u16* t1B    = (u16*)carve((size_t)MLAT * 256 * 2);
  float* resF = (float*)carve((size_t)MLAT * 256 * 4);
  u16* h2B    = (u16*)carve((size_t)MBIG * 256 * 2);

  // --- data-independent precompute: thresholds + weight conversion ---
  row_percentile_kernel<<<NL, 256, 0, stream>>>(thrD, 32, 64, NI, 2.0f);
  row_percentile_kernel<<<NL, 256, 0, stream>>>(thrP, 32, 32, NL, 10.0f);

  head_w_transpose_kernel<<<256, 256, 0, stream>>>(wtD, downw);
  head_w_transpose_kernel<<<256, 256, 0, stream>>>(wtU, upw);
  for (int i = 0; i < NB; ++i)
    head_w_transpose_kernel<<<256, 256, 0, stream>>>(wtP + (size_t)i * 65536,
                                                     paw + (size_t)i * 65536);
  f32_to_bf16_kernel<<<(NB * 65536) / 256, 256, 0, stream>>>(w1B, m1w, NB * 65536);
  f32_to_bf16_kernel<<<(NB * 65536) / 256, 256, 0, stream>>>(w2B, m2w, NB * 65536);
  f32_to_bf16_kernel<<<(NB * 65536) / 256, 256, 0, stream>>>(wrB, rsw, NB * 65536);
  f32_to_bf16_kernel<<<65536 / 256, 256, 0, stream>>>(wd1B, d1w, 65536);

  // --- encoder ---
  encoder_kernel<<<MBIG, 256, 0, stream>>>(hbBig, x, enw, enb);

  // --- down attention (latent <- input) ---
  build_att_kernel<<<dim3(NL, H), 256, 0, stream>>>(attB, downr, thrD, 32, 64, NL, NI, 1);
  gemm_bf16_kernel<2, 128, 64, 4, 2, 2, 2><<<dim3(MBIG / 128, 4, 1), 256, 0, stream>>>(
      hbBig, wtD, nullptr, vtB, nullptr, nullptr, MBIG, 256, 256, 0, NI);
  gemm_bf16_kernel<1, 128, 32, 8, 1, 1, 2><<<dim3(NL / 128, 1, B * H), 256, 0, stream>>>(
      attB, vtB, nullptr, hbLat, nullptr, nullptr, NL, 32, NI, F_GELU | F_OUTB, 0);

  // --- processing blocks ---
  for (int i = 0; i < NB; ++i) {
    build_att_kernel<<<dim3(NL, H), 256, 0, stream>>>(attB, par + i * H, thrP,
                                                      32, 32, NL, NL, 1);
    gemm_bf16_kernel<2, 128, 64, 4, 2, 2, 2><<<dim3(MLAT / 128, 4, 1), 256, 0, stream>>>(
        hbLat, wtP + (size_t)i * 65536, nullptr, vtB, nullptr, nullptr,
        MLAT, 256, 256, 0, NL);
    gemm_bf16_kernel<1, 128, 32, 8, 1, 1, 2><<<dim3(NL / 128, 1, B * H), 256, 0, stream>>>(
        attB, vtB, nullptr, paB, nullptr, nullptr, NL, 32, NL, F_GELU | F_OUTB, 0);
    // mlp1: gelu(pa @ w1^T + b1)
    gemm_bf16_kernel<0, 128, 64, 4, 2, 2, 2><<<dim3(MLAT / 128, 4, 1), 256, 0, stream>>>(
        paB, w1B + (size_t)i * 65536, nullptr, t1B, m1b + i * 256, nullptr,
        MLAT, 256, 256, F_BIAS | F_GELU | F_OUTB, 0);
    // residual: h @ rw^T + rb (f32)
    gemm_bf16_kernel<0, 128, 64, 4, 2, 2, 2><<<dim3(MLAT / 128, 4, 1), 256, 0, stream>>>(
        hbLat, wrB + (size_t)i * 65536, resF, nullptr, rsb + i * 256, nullptr,
        MLAT, 256, 256, F_BIAS | F_OUTF, 0);
    // h = gelu(t1 @ w2^T + b2 + residual)
    gemm_bf16_kernel<0, 128, 64, 4, 2, 2, 2><<<dim3(MLAT / 128, 4, 1), 256, 0, stream>>>(
        t1B, w2B + (size_t)i * 65536, nullptr, hbLat, m2b + i * 256, resF,
        MLAT, 256, 256, F_BIAS | F_ADD | F_GELU | F_OUTB, 0);
  }

  // --- up attention (output <- latent) ---
  build_att_kernel<<<dim3(NI, H), 256, 0, stream>>>(attB, upr, nullptr, 64, 32, NI, NL, 0);
  gemm_bf16_kernel<2, 128, 64, 4, 2, 2, 2><<<dim3(MLAT / 128, 4, 1), 256, 0, stream>>>(
      hbLat, wtU, nullptr, vtB, nullptr, nullptr, MLAT, 256, 256, 0, NL);
  gemm_bf16_kernel<1, 128, 32, 8, 1, 1, 2><<<dim3(NI / 128, 1, B * H), 256, 0, stream>>>(
      attB, vtB, nullptr, hbBig, nullptr, nullptr, NI, 32, NL, F_GELU | F_OUTB, 0);

  // --- decoder ---
  gemm_bf16_kernel<0, 128, 64, 4, 2, 2, 2><<<dim3(MBIG / 128, 4, 1), 256, 0, stream>>>(
      hbBig, wd1B, nullptr, h2B, d1b, nullptr, MBIG, 256, 256,
      F_BIAS | F_GELU | F_OUTB, 0);
  decoder_fc2_kernel<<<MBIG / 8, 256, 0, stream>>>(outp, h2B, d2w, d2b);
}
